// MTFNet_68101001446009
// MI455X (gfx1250) — matrix operand order repaired and checked by reading, not compiled, not run on hardware
//
#include <hip/hip_runtime.h>

typedef __attribute__((ext_vector_type(16))) _Float16 v16h;
typedef __attribute__((ext_vector_type(8)))  float    v8f;

#define NIMG 8
#define CCH  8
#define HH   256
#define HOUT 128
#define PPOS 16384   // 128*128
#define LK   11
#define KK2  121
#define NC968 968    // C*121
#define HP   266     // padded width 256+2*5
#define ROWS 41      // 2*16 + 11 - 2 rows of padded input per 16-output-row stripe

// ---- CDNA5 async memory->LDS staging (guarded: falls back to plain ld/st) ----
#if defined(__gfx1250__) && __has_builtin(__builtin_amdgcn_global_load_async_to_lds_b32) && \
    __has_builtin(__builtin_amdgcn_s_wait_asynccnt)
#define USE_ASYNC_LDS 1
#else
#define USE_ASYNC_LDS 0
#endif

typedef __attribute__((address_space(1))) int* gptr_i32;
typedef __attribute__((address_space(3))) int* lptr_i32;

__device__ __forceinline__ void stage_b32(const float* __restrict__ g, float* l) {
#if USE_ASYNC_LDS
    __builtin_amdgcn_global_load_async_to_lds_b32(
        (gptr_i32)(g), (lptr_i32)(l), 0, 0);
#else
    *l = *g;
#endif
}

__device__ __forceinline__ void stage_wait() {
#if USE_ASYNC_LDS
    __builtin_amdgcn_s_wait_asynccnt(0);
#endif
}

__device__ __forceinline__ int reflect256(int s) {
    if (s < 0) s = -s;
    if (s > 255) s = 510 - s;
    return s;
}

// ---------------- zero helper ----------------
__global__ void zero_f32(float* p, int n) {
    int i = blockIdx.x * 256 + threadIdx.x;
    if (i < n) p[i] = 0.f;
}

// ---------------- pass 1: depthwise blur (stride 2, reflect pad 5) + residual ----------------
__global__ __launch_bounds__(256) void blur_resid(const float* __restrict__ x,
                                                  const float* __restrict__ y,
                                                  const float* __restrict__ kin,
                                                  float* __restrict__ D) {
    const int ht = blockIdx.x, c = blockIdx.y, n = blockIdx.z;
    const int tid = threadIdx.x;
    __shared__ float xs[ROWS * HP];
    __shared__ float ks[KK2];
    const float* xc = x + (n * CCH + c) * (HH * HH);
    __builtin_prefetch(xc + ht * 32 * HH, 0, 0);   // global_prefetch_b8
    const int row0 = ht * 32;                      // padded row base = 2*ho0
    for (int t = tid; t < ROWS * HP; t += 256) {
        int r = t / HP, q = t - r * HP;
        int sr = reflect256(row0 + r - 5);
        int sc = reflect256(q - 5);
        stage_b32(&xc[sr * HH + sc], &xs[t]);      // async mem->LDS on CDNA5
    }
    if (tid < KK2) ks[tid] = kin[(n * CCH + c) * KK2 + tid];
    stage_wait();
    __syncthreads();

    const int wo = tid & 127, lr0 = tid >> 7;
    for (int lr = lr0; lr < 16; lr += 2) {
        float acc = 0.f;
        const float* row = &xs[(2 * lr) * HP + 2 * wo];
        for (int i = 0; i < LK; ++i)
            for (int j = 0; j < LK; ++j)
                acc = fmaf(row[i * HP + j], ks[i * LK + j], acc);
        int p = (ht * 16 + lr) * HOUT + wo;
        int idx = (n * CCH + c) * PPOS + p;
        D[idx] = acc - y[idx];
    }
}

// ---------------- pass 2: G[n,k,c] += sum_p x_col[n,c,k,p] * Dflat[n, p*8+c] ----------------
__global__ __launch_bounds__(256) void g_accum(const float* __restrict__ x,
                                               const float* __restrict__ D,
                                               float* __restrict__ G) {
    const int ht = blockIdx.x, cc = blockIdx.y, n = blockIdx.z;
    const int tid = threadIdx.x;
    __shared__ float xs[ROWS * HP];
    __shared__ float rb[2048];
    const float* xc = x + (n * CCH + cc) * (HH * HH);
    const int row0 = ht * 32;
    for (int t = tid; t < ROWS * HP; t += 256) {
        int r = t / HP, q = t - r * HP;
        int sr = reflect256(row0 + r - 5);
        int sc = reflect256(q - 5);
        stage_b32(&xc[sr * HH + sc], &xs[t]);
    }
    // torch-view gather: R[n,p,c] = D[n].flat[p*8 + c]
    const float* Dn = D + n * (CCH * PPOS);
    for (int t = tid; t < 2048; t += 256) {
        int lr = t >> 7, wo = t & 127;
        int p = (ht * 16 + lr) * HOUT + wo;
        stage_b32(&Dn[p * CCH + cc], &rb[t]);
    }
    stage_wait();
    __syncthreads();

    const int k2 = tid >> 1, half = tid & 1;
    if (k2 < KK2) {
        const int i = k2 / LK, j = k2 - i * LK;
        float acc = 0.f;
        for (int lr = half * 8; lr < half * 8 + 8; ++lr) {
            const float* row = &xs[(2 * lr + i) * HP + j];
            const float* rr  = &rb[lr * HOUT];
            for (int wo = 0; wo < HOUT; ++wo)
                acc = fmaf(row[2 * wo], rr[wo], acc);
        }
        atomicAdd(&G[n * NC968 + k2 * CCH + cc], acc);
    }
}

// ---------------- G_K = k - (gamma/10) * G/(h*w)  (flat views collapse to identity) ----------------
__global__ void gk_make(const float* __restrict__ kin, const float* __restrict__ G,
                        const float* __restrict__ gamma, float* __restrict__ GK) {
    int i = blockIdx.x * 256 + threadIdx.x;
    if (i < NIMG * NC968)
        GK[i] = kin[i] - gamma[0] * (0.1f / 16384.f) * G[i];
}

// ---------------- ProNet 3x3 conv (C=8->8, zero pad 1, 11x11) as WMMA im2col GEMM ----------------
// One wave per image. M=16 (8 out-ch padded), K=96 (72 padded, 3 chunks of 32), N=8 tiles of 16 pos.
__global__ __launch_bounds__(32) void conv3x3_wmma(const float* __restrict__ in,
                                                   const float* __restrict__ w,
                                                   const float* __restrict__ b,
                                                   float* __restrict__ out,
                                                   int do_relu) {
    const int n = blockIdx.x;
    const int lane = threadIdx.x;
    __shared__ _Float16 inpad[CCH * 169];  // 8 ch x 13x13 zero-padded
    __shared__ float bias[CCH];

    for (int t = lane; t < CCH * 169; t += 32) {
        int ic = t / 169, rem = t - ic * 169;
        int r = rem / 13, cidx = rem - r * 13;
        float v = 0.f;
        if (r >= 1 && r <= 11 && cidx >= 1 && cidx <= 11)
            v = in[n * NC968 + ic * KK2 + (r - 1) * LK + (cidx - 1)];
        inpad[t] = (_Float16)v;
    }
    if (lane < CCH) bias[lane] = b[lane];
    __syncthreads();

    const int M = lane & 15;
    const int lane_hi = lane >> 4;

    // A fragment: weights, row M = out-channel, K = ic*9 + di*3 + dj
    v16h afrag[3];
    for (int ch = 0; ch < 3; ++ch) {
        v16h a;
        for (int e = 0; e < 16; ++e) {
            int K = ch * 32 + (e & 7) + ((e >> 3) << 4) + (lane_hi << 3);
            float v = (M < CCH && K < 72) ? w[M * 72 + K] : 0.f;
            a[e] = (_Float16)v;
        }
        afrag[ch] = a;
    }

    const int Ncol = lane & 15;
    for (int tile = 0; tile < 8; ++tile) {
        const int p = tile * 16 + Ncol;
        const int pv = (p < KK2);
        const int pc = pv ? p : 0;
        const int pi = pc / LK, pj = pc - pi * LK;
        v8f c = {};
        for (int ch = 0; ch < 3; ++ch) {
            v16h bf;
            for (int e = 0; e < 16; ++e) {
                int K = ch * 32 + (e & 7) + ((e >> 3) << 4) + (lane_hi << 3);
                int kc = (K < 72) ? K : 0;
                int ic = kc / 9, rem = kc - ic * 9;
                int di = rem / 3, dj = rem - di * 3;
                _Float16 v = inpad[ic * 169 + (pi + di) * 13 + (pj + dj)];
                bf[e] = (pv && K < 72) ? v : (_Float16)0;
            }
            c = __builtin_amdgcn_wmma_f32_16x16x32_f16(
                    false, afrag[ch], false, bf, (short)0, c, false, false);
        }
        if (lane < 16 && p < KK2) {
            for (int e = 0; e < CCH; ++e) {
                float v = c[e] + bias[e];
                if (do_relu) v = fmaxf(v, 0.f);
                out[n * NC968 + e * KK2 + p] = v;
            }
        }
    }
}

// ---------------- residual combine: out = relu(a + 0.1*t) ----------------
__global__ void resid_relu(const float* __restrict__ a, const float* __restrict__ t,
                           float* __restrict__ o) {
    int i = blockIdx.x * 256 + threadIdx.x;
    if (i < NIMG * NC968) o[i] = fmaxf(fmaf(0.1f, t[i], a[i]), 0.f);
}

// ---------------- per-(n,c) L1 normalize ----------------
__global__ __launch_bounds__(128) void norm_k(const float* __restrict__ K, float* __restrict__ out) {
    const int bb = blockIdx.x, tid = threadIdx.x;
    __shared__ float s[128];
    float v = (tid < KK2) ? K[bb * KK2 + tid] : 0.f;
    s[tid] = v;
    __syncthreads();
    for (int off = 64; off > 0; off >>= 1) {
        if (tid < off) s[tid] += s[tid + off];
        __syncthreads();
    }
    float sum = s[0];
    if (tid < KK2) out[bb * KK2 + tid] = v / sum;
}

extern "C" void kernel_launch(void* const* d_in, const int* in_sizes, int n_in,
                              void* d_out, int out_size, void* d_ws, size_t ws_size,
                              hipStream_t stream) {
    const float* x     = (const float*)d_in[0];   // [8,8,256,256]
    const float* y     = (const float*)d_in[1];   // [8,8,128,128]
    const float* kin   = (const float*)d_in[2];   // [8,8,11,11]
    // d_in[3] = sf (compile-time constant 2)
    const float* gamma = (const float*)d_in[4];   // scalar
    const float* W[9];
    const float* B[9];
    for (int i = 0; i < 9; ++i) {
        W[i] = (const float*)d_in[5 + 2 * i];
        B[i] = (const float*)d_in[6 + 2 * i];
    }

    float* ws = (float*)d_ws;
    float* D  = ws;                       // 1,048,576 f32
    float* G  = D  + NIMG * CCH * PPOS;   // 7744
    float* GK = G  + NIMG * NC968;        // 7744
    float* T0 = GK + NIMG * NC968;        // 7744
    float* T1 = T0 + NIMG * NC968;        // 7744
    float* XA = T1 + NIMG * NC968;        // 7744
    float* XB = XA + NIMG * NC968;        // 7744
    float* KB = XB + NIMG * NC968;        // 7744
    float* out = (float*)d_out;

    const int NE = NIMG * NC968;          // 7744
    const int EB = (NE + 255) / 256;

    zero_f32<<<EB, 256, 0, stream>>>(G, NE);
    blur_resid<<<dim3(8, CCH, NIMG), 256, 0, stream>>>(x, y, kin, D);
    g_accum  <<<dim3(8, CCH, NIMG), 256, 0, stream>>>(x, D, G);
    gk_make  <<<EB, 256, 0, stream>>>(kin, G, gamma, GK);

    const float* xin = GK;
    float* xbufs[2] = {XA, XB};
    for (int bl = 0; bl < 4; ++bl) {
        conv3x3_wmma<<<NIMG, 32, 0, stream>>>(xin, W[2 * bl],     B[2 * bl],     T0, 1);
        conv3x3_wmma<<<NIMG, 32, 0, stream>>>(T0,  W[2 * bl + 1], B[2 * bl + 1], T1, 0);
        float* xo = xbufs[bl & 1];
        resid_relu<<<EB, 256, 0, stream>>>(xin, T1, xo);
        xin = xo;
    }
    conv3x3_wmma<<<NIMG, 32, 0, stream>>>(xin, W[8], B[8], T0, 0);
    resid_relu<<<EB, 256, 0, stream>>>(GK, T0, KB);   // K = relu(g + 0.1*conv5(x4)); outer relu idempotent
    norm_k<<<NIMG * CCH, 128, 0, stream>>>(KB, out);
}
